// BiLSTM_CRF_55216099558135
// MI455X (gfx1250) — matrix operand order repaired and checked by reading, not compiled, not run on hardware
//
#include <hip/hip_runtime.h>
#include <hip/hip_bf16.h>
#include <string.h>

// ---------------- types ----------------
typedef __attribute__((ext_vector_type(16))) __bf16 v16bf;
typedef __attribute__((ext_vector_type(8)))  float  v8f;
typedef __attribute__((ext_vector_type(4)))  unsigned int u32x4;
typedef __attribute__((ext_vector_type(8)))  int i32x8;
typedef __attribute__((ext_vector_type(4)))  int i32x4;

union Frag {
    v16bf bf;
    u32x4 q[2];
};

#define SEQ 2048
#define EDIM 512
#define HDIM 512
#define GDIM 2048            // 4*H gates
#define NTAGS 81
#define START_TAG 79
#define STOP_TAG 80
#define NEGV (-10000.0f)

// ---- TDM availability / arity hedge ----
#if defined(__has_builtin)
#if __has_builtin(__builtin_amdgcn_tensor_load_to_lds)
#define HAVE_TDM 1
#endif
#endif
#ifndef HAVE_TDM
#define HAVE_TDM 0
#endif

#if HAVE_TDM
// Issue a TDM 1-row copy: GDIM f32 elements from gsrc (global) to ldst (LDS).
// D# per CDNA5 ISA ch.8: group0 {count=1, lds_addr, global_addr, type=2},
// group1 {data_size=2(4B), tensor_dim0=2048, tensor_dim1=1, tile_dim0=2048,
//         tile_dim1=1, tensor_dim0_stride=2048}.
static __device__ __forceinline__ void tdm_load_row(const float* gsrc, float* ldst) {
    unsigned long long ga = (unsigned long long)(uintptr_t)gsrc;
    unsigned int lds = (unsigned int)(uintptr_t)ldst;   // low 32 bits = LDS byte addr
    u32x4 g0;
    g0[0] = 1u;                                          // count=1
    g0[1] = lds;                                         // lds_addr
    g0[2] = (unsigned int)ga;                            // global_addr[31:0]
    g0[3] = (unsigned int)((ga >> 32) & 0x1FFFFFFu) | (2u << 30);  // gaddr hi | type=2
    i32x8 g1;
    g1[0] = 0x00020000;                 // workgroup_mask=0, data_size=2 (4 bytes)
    g1[1] = (GDIM & 0xffff) << 16;      // tensor_dim0 lo16 (abar addr = 0)
    g1[2] = (GDIM >> 16) | (1 << 16);   // tensor_dim0 hi16 | tensor_dim1 lo16 = 1
    g1[3] = (GDIM << 16);               // tensor_dim1 hi16 = 0 | tile_dim0 = 2048
    g1[4] = 1;                          // tile_dim1 = 1, tile_dim2 = 0
    g1[5] = GDIM;                       // tensor_dim0_stride lo32
    g1[6] = 0;
    g1[7] = 0;
    i32x4 z4 = (i32x4)0;
#if __has_include(<hip/amd_detail/amd_gfx1250_TDM.h>)
    i32x8 z8 = (i32x8)0;
    __builtin_amdgcn_tensor_load_to_lds(g0, g1, z4, z4, z8, 0);   // 6-arg toolchain
#else
    __builtin_amdgcn_tensor_load_to_lds(g0, g1, z4, z4, 0);       // 5-arg toolchain
#endif
}
#endif

// f32 -> bf16 (round to nearest even)
static __device__ __forceinline__ unsigned short f2bf(float f) {
    unsigned int u = __builtin_bit_cast(unsigned int, f);
    unsigned int lsb = (u >> 16) & 1u;
    u += 0x7fffu + lsb;
    return (unsigned short)(u >> 16);
}

static __device__ __forceinline__ float sigm(float x) {
    return 1.0f / (1.0f + __expf(-x));
}

// ---------------- pack kernels ----------------
__global__ void pack_bf16_kernel(const float* __restrict__ src,
                                 unsigned short* __restrict__ dst, int n) {
    int i = blockIdx.x * blockDim.x + threadIdx.x;
    if (i < n) dst[i] = f2bf(src[i]);
}

// gather x[t] = emb[sentence[t]] and convert to bf16
__global__ void gather_x_bf16_kernel(const int* __restrict__ sent,
                                     const float* __restrict__ emb,
                                     unsigned short* __restrict__ xbf) {
    int t = blockIdx.x;
    int e = threadIdx.x;                       // blockDim == 512
    int tok = sent[t];
    xbf[t * EDIM + e] = f2bf(emb[(long)tok * EDIM + e]);
}

// ---------------- input projection GEMM (WMMA bf16) ----------------
// G[t, g] = sum_e X[t,e] * W[g,e] + bih[g] + bhh[g]
// One wave computes a 16x64 strip (1 M-tile x 4 N-tiles): the A fragment is
// loaded once per K step and reused across 4 WMMAs.
__global__ void __launch_bounds__(256)
input_proj_wmma(const unsigned short* __restrict__ Xbf,   // SEQ x EDIM
                const unsigned short* __restrict__ Wbf,   // GDIM x EDIM
                const float* __restrict__ bih,
                const float* __restrict__ bhh,
                float* __restrict__ Gout)                 // SEQ x GDIM
{
    const int lane = threadIdx.x & 31;
    const int wave = threadIdx.x >> 5;
    const int ntb = (blockIdx.x * 8 + wave) * 4;   // first of 4 N tiles
    const int m0 = blockIdx.y * 16;

    const __bf16* X = (const __bf16*)Xbf;
    const __bf16* W = (const __bf16*)Wbf;

    // A fragment: row m = m0 + (lane&15); K halves at +8*(lane>=16), +16 more
    const int mA = m0 + (lane & 15);
    const int kA = (lane >> 4) << 3;        // 0 or 8
    // B fragment: col n = n0 + (lane&15); K halves at +8*(lane>=16), +16 more
    const int nB = ntb * 16 + (lane & 15);
    const int kB = (lane >> 4) << 3;        // 0 or 8

    const __bf16* pa = X + (long)mA * EDIM + kA;
    const __bf16* pb = W + (long)nB * EDIM + kB;

    v8f acc[4];
#pragma unroll
    for (int j = 0; j < 4; ++j) acc[j] = (v8f)0.0f;

    for (int k = 0; k < EDIM; k += 32) {
        __builtin_prefetch(pa + k + 128, 0, 1);   // global_prefetch
        Frag a;
        a.q[0] = *(const u32x4*)(pa + k);
        a.q[1] = *(const u32x4*)(pa + k + 16);
#pragma unroll
        for (int j = 0; j < 4; ++j) {
            const __bf16* pbj = pb + (long)j * 16 * EDIM;
            Frag b;
            b.q[0] = *(const u32x4*)(pbj + k);
            b.q[1] = *(const u32x4*)(pbj + k + 16);
            acc[j] = __builtin_amdgcn_wmma_f32_16x16x32_bf16(
                false, a.bf, false, b.bf, (short)0, acc[j], false, false);
        }
    }

    const int mrow = m0 + ((lane >> 4) << 3);
#pragma unroll
    for (int j = 0; j < 4; ++j) {
        const int n = (ntb + j) * 16 + (lane & 15);
        const float bias = bih[n] + bhh[n];
#pragma unroll
        for (int r = 0; r < 8; ++r)
            Gout[(long)(mrow + r) * GDIM + n] = acc[j][r] + bias;
    }
}

// ---------------- sequential LSTM recurrence (WMMA GEMV + TDM prefetch) -----
// blockIdx.x = direction (0 fwd, 1 bwd). 1024 threads = 32 waves.
// Per step: gates = Whh @ h via WMMA with h broadcast across all 16 B columns.
// While the WMMA phase runs, wave 0 TDM-prefetches the next step's input-
// projection row G[t+1] (8KB) into an LDS double buffer (TENSORcnt tracked).
__global__ void __launch_bounds__(1024)
lstm_recur(const unsigned short* __restrict__ WhhF,
           const unsigned short* __restrict__ WhhB,
           const float* __restrict__ Gf,
           const float* __restrict__ Gb,
           const float* __restrict__ h0,
           const float* __restrict__ c0,
           float* __restrict__ hf,
           float* __restrict__ hb)
{
    __shared__ __align__(32) unsigned short h_bf[HDIM];
    __shared__ float c_s[HDIM];
    __shared__ float gates[GDIM];
#if HAVE_TDM
    __shared__ __align__(16) float gbuf[2][GDIM];
#endif

    const int dir = blockIdx.x;
    const __bf16* W = (const __bf16*)(dir ? WhhB : WhhF);
    const float*  G = dir ? Gb : Gf;
    float*       hs = dir ? hb : hf;

    const int tid = threadIdx.x;
    const int lane = tid & 31;
    const int wave = tid >> 5;

    if (tid < HDIM) {
        h_bf[tid] = f2bf(h0[dir * HDIM + tid]);
        c_s[tid]  = c0[dir * HDIM + tid];
    }
#if HAVE_TDM
    if (wave == 0) {   // prime buffer 0 with first step's G row
        tdm_load_row(G + (long)(dir ? SEQ - 1 : 0) * GDIM, &gbuf[0][0]);
    }
#endif
    __syncthreads();

    const int mA = lane & 15;
    const int kA = (lane >> 4) << 3;    // 0 or 8
    const int kB = (lane >> 4) << 3;    // 0 or 8

    for (int s = 0; s < SEQ; ++s) {
        const int t = dir ? (SEQ - 1 - s) : s;

#if HAVE_TDM
        // kick off next step's G row while we compute this step's GEMV
        if (wave == 0) {
            const int sn = (s + 1 < SEQ) ? (s + 1) : s;   // clamp last iter
            const int tn = dir ? (SEQ - 1 - sn) : sn;
            tdm_load_row(G + (long)tn * GDIM, &gbuf[(s + 1) & 1][0]);
        }
#endif

        // ---- gates = Whh @ h (2048x512 GEMV), 4 row-tiles per wave ----
#pragma unroll
        for (int qq = 0; qq < 4; ++qq) {
            const int m0 = (wave * 4 + qq) * 16;
            const __bf16* pa = W + (long)(m0 + mA) * HDIM + kA;
            v8f acc = (v8f)0.0f;
            for (int k = 0; k < HDIM; k += 32) {
                Frag a, b;
                a.q[0] = *(const u32x4*)(pa + k);
                a.q[1] = *(const u32x4*)(pa + k + 16);
                const __bf16* pb = (const __bf16*)&h_bf[k + kB]; // LDS broadcast
                b.q[0] = *(const u32x4*)(pb);
                b.q[1] = *(const u32x4*)(pb + 16);
                acc = __builtin_amdgcn_wmma_f32_16x16x32_bf16(
                    false, a.bf, false, b.bf, (short)0, acc, false, false);
            }
            if ((lane & 15) == 0) {             // column 0 holders: lanes 0 & 16
                const int mbase = m0 + ((lane >> 4) << 3);
#pragma unroll
                for (int r = 0; r < 8; ++r) gates[mbase + r] = acc[r];
            }
        }

#if HAVE_TDM
        // retire the older TDM transfer (this step's buffer); transfers are
        // in-order per wave, so TENSORcnt<=1 means gbuf[s&1] is complete.
        if (wave == 0) __builtin_amdgcn_s_wait_tensorcnt(1);
#endif
        __syncthreads();

        // ---- elementwise LSTM cell update ----
        if (tid < HDIM) {
#if HAVE_TDM
            const float* gt = &gbuf[s & 1][0];
#else
            const float* gt = G + (long)t * GDIM;
#endif
            float gi = gates[tid]            + gt[tid];
            float gf = gates[HDIM + tid]     + gt[HDIM + tid];
            float gg = gates[2 * HDIM + tid] + gt[2 * HDIM + tid];
            float go = gates[3 * HDIM + tid] + gt[3 * HDIM + tid];
            float c2 = sigm(gf) * c_s[tid] + sigm(gi) * tanhf(gg);
            float h2 = sigm(go) * tanhf(c2);
            c_s[tid] = c2;
            h_bf[tid] = f2bf(h2);
            hs[(long)t * HDIM + tid] = h2;
        }
        __syncthreads();
    }
}

// ---------------- feature projection (small, fp32 VALU) ----------------
__global__ void __launch_bounds__(128)
feats_kernel(const float* __restrict__ hf, const float* __restrict__ hb,
             const float* __restrict__ Wt, const float* __restrict__ bt,
             float* __restrict__ feats)
{
    __shared__ float hrow[2 * HDIM];
    const int t = blockIdx.x;
    for (int i = threadIdx.x; i < HDIM; i += 128) {
        hrow[i]        = hf[(long)t * HDIM + i];
        hrow[HDIM + i] = hb[(long)t * HDIM + i];
    }
    __syncthreads();
    const int k = threadIdx.x;
    if (k < NTAGS) {
        const float* w = Wt + (long)k * (2 * HDIM);
        float acc = bt[k];
        for (int j = 0; j < 2 * HDIM; ++j) acc = fmaf(hrow[j], w[j], acc);
        feats[(long)t * NTAGS + k] = acc;
    }
}

// ---------------- Viterbi decode + backtrack ----------------
__global__ void __launch_bounds__(128)
viterbi_kernel(const float* __restrict__ feats,
               const float* __restrict__ trans,
               int* __restrict__ bps, float* __restrict__ stepvars,
               float* __restrict__ out)
{
    __shared__ float tr[NTAGS * NTAGS];
    __shared__ float fv[NTAGS];
    __shared__ float fv2[NTAGS];
    __shared__ float term[NTAGS];
    __shared__ int   bestIdx;

    const int tid = threadIdx.x;
    for (int i = tid; i < NTAGS * NTAGS; i += 128) tr[i] = trans[i];
    if (tid < NTAGS) fv[tid] = (tid == START_TAG) ? 0.0f : NEGV;
    __syncthreads();

    for (int t = 0; t < SEQ; ++t) {
        if (tid < NTAGS) {
            const float* tri = &tr[tid * NTAGS];
            float best = -3.402823466e38f;
            int bj = 0;
            for (int j = 0; j < NTAGS; ++j) {
                float v = fv[j] + tri[j];
                if (v > best) { best = v; bj = j; }   // first-max == jnp.argmax
            }
            float v2 = best + feats[(long)t * NTAGS + tid];
            fv2[tid] = v2;
            bps[(long)t * NTAGS + tid] = bj;
            stepvars[(long)t * NTAGS + tid] = v2;
        }
        __syncthreads();
        if (tid < NTAGS) fv[tid] = fv2[tid];
        __syncthreads();
    }

    if (tid < NTAGS) term[tid] = fv[tid] + tr[STOP_TAG * NTAGS + tid];
    __syncthreads();
    if (tid == 0) {
        float bv = -3.402823466e38f;
        int bi = 0;
        for (int j = 0; j < NTAGS; ++j)
            if (term[j] > bv) { bv = term[j]; bi = j; }
        out[0] = bv;                                  // path_score
        int tag = bi;
        for (int t = SEQ - 1; t >= 0; --t) {          // reverse scan backtrack
            out[1 + t] = (float)tag;
            tag = bps[(long)t * NTAGS + tag];
        }
        bestIdx = bi;
    }
    __syncthreads();
    for (int i = tid; i < SEQ; i += 128)              // step_vars[::-1, best]
        out[1 + SEQ + i] = stepvars[(long)(SEQ - 1 - i) * NTAGS + bestIdx];
}

// ---------------- launch ----------------
extern "C" void kernel_launch(void* const* d_in, const int* in_sizes, int n_in,
                              void* d_out, int out_size, void* d_ws, size_t ws_size,
                              hipStream_t stream) {
    const int*   sent  = (const int*)  d_in[0];
    const float* h0    = (const float*)d_in[1];
    const float* c0    = (const float*)d_in[2];
    const float* emb   = (const float*)d_in[3];
    const float* Wih_f = (const float*)d_in[4];
    const float* Whh_f = (const float*)d_in[5];
    const float* bih_f = (const float*)d_in[6];
    const float* bhh_f = (const float*)d_in[7];
    const float* Wih_b = (const float*)d_in[8];
    const float* Whh_b = (const float*)d_in[9];
    const float* bih_b = (const float*)d_in[10];
    const float* bhh_b = (const float*)d_in[11];
    const float* Wt    = (const float*)d_in[12];
    const float* bt    = (const float*)d_in[13];
    const float* trans = (const float*)d_in[14];

    // workspace carve-up (256B aligned slabs)
    char* ws = (char*)d_ws;
    size_t off = 0;
    auto alloc = [&](size_t bytes) {
        char* p = ws + off;
        off += (bytes + 255) & ~(size_t)255;
        return p;
    };
    unsigned short* Xbf    = (unsigned short*)alloc((size_t)SEQ * EDIM * 2);
    unsigned short* WihFbf = (unsigned short*)alloc((size_t)GDIM * EDIM * 2);
    unsigned short* WihBbf = (unsigned short*)alloc((size_t)GDIM * EDIM * 2);
    unsigned short* WhhFbf = (unsigned short*)alloc((size_t)GDIM * HDIM * 2);
    unsigned short* WhhBbf = (unsigned short*)alloc((size_t)GDIM * HDIM * 2);
    float* Gf       = (float*)alloc((size_t)SEQ * GDIM * 4);
    float* Gb       = (float*)alloc((size_t)SEQ * GDIM * 4);
    float* hfbuf    = (float*)alloc((size_t)SEQ * HDIM * 4);
    float* hbbuf    = (float*)alloc((size_t)SEQ * HDIM * 4);
    float* featsbuf = (float*)alloc((size_t)SEQ * NTAGS * 4);
    int*   bpsbuf   = (int*)  alloc((size_t)SEQ * NTAGS * 4);
    float* svbuf    = (float*)alloc((size_t)SEQ * NTAGS * 4);
    (void)ws_size; (void)n_in; (void)in_sizes; (void)out_size;

    // 1) pack inputs to bf16
    gather_x_bf16_kernel<<<SEQ, EDIM, 0, stream>>>(sent, emb, Xbf);
    const int NW = GDIM * EDIM;
    pack_bf16_kernel<<<(NW + 255) / 256, 256, 0, stream>>>(Wih_f, WihFbf, NW);
    pack_bf16_kernel<<<(NW + 255) / 256, 256, 0, stream>>>(Wih_b, WihBbf, NW);
    pack_bf16_kernel<<<(NW + 255) / 256, 256, 0, stream>>>(Whh_f, WhhFbf, NW);
    pack_bf16_kernel<<<(NW + 255) / 256, 256, 0, stream>>>(Whh_b, WhhBbf, NW);

    // 2) input projections, WMMA GEMM (both directions)
    dim3 ggrid(GDIM / 16 / 32, SEQ / 16);   // (4, 128) blocks, 8 waves x 4 N-tiles
    input_proj_wmma<<<ggrid, 256, 0, stream>>>(Xbf, WihFbf, bih_f, bhh_f, Gf);
    input_proj_wmma<<<ggrid, 256, 0, stream>>>(Xbf, WihBbf, bih_b, bhh_b, Gb);

    // 3) sequential recurrence, fwd+bwd in parallel (WMMA GEMV + TDM prefetch)
    lstm_recur<<<2, 1024, 0, stream>>>(WhhFbf, WhhBbf, Gf, Gb, h0, c0, hfbuf, hbbuf);

    // 4) tag features
    feats_kernel<<<SEQ, 128, 0, stream>>>(hfbuf, hbbuf, Wt, bt, featsbuf);

    // 5) Viterbi + backtrack -> d_out = [score | path(2048) | best_tag_score(2048)]
    viterbi_kernel<<<1, 128, 0, stream>>>(featsbuf, trans, bpsbuf, svbuf, (float*)d_out);
}
